// BMN_19920058319502
// MI455X (gfx1250) — compile-verified, matrix operand order checked by reading, not run on hardware
//
#include <hip/hip_runtime.h>
#include <hip/hip_bf16.h>
#include <stdint.h>

// ---------------------------------------------------------------------------
// BMN forward for MI455X (gfx1250, wave32, WMMA).
//
//  * One strided/batched WMMA GEMM template handles all GEMM-shaped work:
//    fp32 global -> f16 LDS tiles -> 4x v_wmma_f32_16x16x32_f16 per wave per
//    K-step (64x128 block tile, 8 waves each computing 32x32).
//    Interior blocks take a guard-free straight-line staging path with
//    coalesced (stride-specialized) loads; boundary blocks take a clamped
//    slow path. K must be a multiple of 32 (true for every call site).
//  * bm matmul + conv3d fused: p = (c3_w x base2) x repacked_mask; kills the
//    642 MB bm intermediate and 41 GMAC of conv3d work. Repacked mask
//    (124 MB) stays resident in the 192 MB L2 across M-tiles.
//  * Latency-bound pieces (attention T<=99, layernorm, grouped conv1d,
//    sigmoid heads) are plain VALU kernels.
// ---------------------------------------------------------------------------

#define T_SCALE 99
#define CCH 256
#define NSB 32
#define TT (T_SCALE * T_SCALE)          // 9801
#define MASK_COLS (NSB * TT)            // 313632

typedef _Float16 v16h __attribute__((ext_vector_type(16)));
typedef float    v8f  __attribute__((ext_vector_type(8)));

#define BM 64
#define BN 128
#define BK 32
#define LDA_S 40    // halves per LDS row (padded, even)
#define LDB_S 40

union HPack { _Float16 h[2]; unsigned u; };

// C[M,N] = act( A*B + bias + resid ), A(m,k) at A[m*sAm + k*sAk],
// B(k,n) at B[k*sBk + n*sBn], C row-major with ldc. Batched via blockIdx.z:
// zr = z % zdiv, zq = z / zdiv; pointers offset by (zr,zq) strides.
// AK1: sAk == 1 (k contiguous in A).  BK1: sBk == 1 (k contiguous in B).
template <int AK1, int BK1>
__global__ __launch_bounds__(256)
void gemm_wmma_f16(const float* __restrict__ A, long sAm, long sAk,
                   const float* __restrict__ B, long sBk, long sBn,
                   float* __restrict__ C, long ldc,
                   int M, int N, int K,
                   const float* __restrict__ bias, int biasMode,   // 0 none,1 per-N,2 per-M
                   const float* __restrict__ resid, int act,       // act: 0 none, 1 relu
                   int zdiv, long aZr, long aZq, long bZr, long bZq,
                   long cZr, long cZq)
{
    __shared__ _Float16 As[BM * LDA_S];
    __shared__ _Float16 Bs[BN * LDB_S];

    const int z  = blockIdx.z;
    const int zr = z % zdiv;
    const int zq = z / zdiv;
    A += (long)zr * aZr + (long)zq * aZq;
    B += (long)zr * bZr + (long)zq * bZq;
    C += (long)zr * cZr + (long)zq * cZq;
    if (resid) resid += (long)zr * cZr + (long)zq * cZq;

    const int tid  = threadIdx.x;
    const int wid  = tid >> 5;        // wave 0..7
    const int lane = tid & 31;
    const int wm   = wid >> 2;        // 0..1  (M sub-tile, 32 rows each)
    const int wn   = wid & 3;         // 0..3  (N sub-tile, 32 cols each)
    const int hi   = lane >> 4;       // lane half
    const int lm   = lane & 15;

    const int m0 = blockIdx.y * BM;
    const int n0 = blockIdx.x * BN;

    // block-uniform: interior tile needs no guards (K % BK == 0 by contract)
    const bool full = (m0 + BM <= M) && (n0 + BN <= N);

    v8f acc[2][2] = {};

    for (int k0 = 0; k0 < K; k0 += BK) {
        if (full) {
            // ---------- guard-free interior staging ----------
#pragma unroll
            for (int i = 0; i < 4; ++i) {       // A: 1024 k-pairs, 4/thread
                int p  = tid + i * 256;
                int m  = p >> 4;
                int kp = (p & 15) << 1;
                float v0, v1;
                if (AK1) {
                    float2 f = *(const float2*)&A[(long)(m0 + m) * sAm + (k0 + kp)];
                    v0 = f.x; v1 = f.y;
                } else {
                    const float* Ar = A + (long)(m0 + m) * sAm;
                    v0 = Ar[(long)(k0 + kp) * sAk];
                    v1 = Ar[(long)(k0 + kp + 1) * sAk];
                }
                HPack pk;
                pk.h[0] = (_Float16)v0;
                pk.h[1] = (_Float16)v1;
                *(unsigned*)&As[m * LDA_S + kp] = pk.u;
            }
            if (BK1) {
#pragma unroll
                for (int i = 0; i < 8; ++i) {   // B: 2048 k-pairs, 8/thread
                    int p  = tid + i * 256;
                    int n  = p >> 4;
                    int kp = (p & 15) << 1;
                    float2 f = *(const float2*)&B[(long)(n0 + n) * sBn + (k0 + kp)];
                    HPack pk;
                    pk.h[0] = (_Float16)f.x;
                    pk.h[1] = (_Float16)f.y;
                    *(unsigned*)&Bs[n * LDB_S + kp] = pk.u;
                }
            } else {
                // n contiguous in memory: lanes walk n for coalescing
#pragma unroll
                for (int i = 0; i < 16; ++i) {  // 4096 elems, 16/thread
                    int p = tid + i * 256;
                    int k = p >> 7;
                    int n = p & 127;
                    float v = B[(long)(k0 + k) * sBk + (long)(n0 + n) * sBn];
                    Bs[n * LDB_S + k] = (_Float16)v;
                }
            }
            // prefetch next K-tile (global_prefetch_b8 on gfx1250)
            if (k0 + BK < K) {
                __builtin_prefetch(&A[(long)(m0 + (tid & 63)) * sAm +
                                      (long)(k0 + BK) * sAk], 0, 1);
                __builtin_prefetch(&B[(long)(k0 + BK) * sBk +
                                      (long)(n0 + (tid & 127)) * sBn], 0, 1);
            }
        } else {
            // ---------- clamped boundary staging ----------
#pragma unroll
            for (int i = 0; i < 4; ++i) {
                int p  = tid + i * 256;
                int m  = p >> 4;
                int kp = (p & 15) << 1;
                int gm = m0 + m, gk = k0 + kp;
                int cmi = gm < M ? gm : M - 1;
                const float* Ar = A + (long)cmi * sAm;
                float v0 = Ar[(long)gk * sAk];
                float v1 = Ar[(long)(gk + 1) * sAk];
                v0 = (gm < M) ? v0 : 0.f;
                v1 = (gm < M) ? v1 : 0.f;
                HPack pk;
                pk.h[0] = (_Float16)v0;
                pk.h[1] = (_Float16)v1;
                *(unsigned*)&As[m * LDA_S + kp] = pk.u;
            }
#pragma unroll
            for (int i = 0; i < 16; ++i) {
                int p = tid + i * 256;
                int k = p >> 7;
                int n = p & 127;
                int gn = n0 + n;
                int cni = gn < N ? gn : N - 1;
                float v = B[(long)(k0 + k) * sBk + (long)cni * sBn];
                v = (gn < N) ? v : 0.f;
                Bs[n * LDB_S + k] = (_Float16)v;
            }
        }
        __syncthreads();

        // ---- fragments per gfx1250 16-bit A/B layout ----
        // lane = 16*hi + (m|n); half h -> k = (h&7) + 16*(h>>3) + 8*hi
        v16h af[2], bf[2];
#pragma unroll
        for (int mi = 0; mi < 2; ++mi) {
            const _Float16* Ar = &As[(wm * 32 + mi * 16 + lm) * LDA_S];
#pragma unroll
            for (int h = 0; h < 16; ++h) {
                int k = (h & 7) + ((h >> 3) << 4) + (hi << 3);
                af[mi][h] = Ar[k];
            }
        }
#pragma unroll
        for (int ni = 0; ni < 2; ++ni) {
            const _Float16* Br = &Bs[(wn * 32 + ni * 16 + lm) * LDB_S];
#pragma unroll
            for (int h = 0; h < 16; ++h) {
                int k = (h & 7) + ((h >> 3) << 4) + (hi << 3);
                bf[ni][h] = Br[k];
            }
        }
        // 4 WMMAs per K-step: A fragments reused across both B fragments
#pragma unroll
        for (int mi = 0; mi < 2; ++mi)
#pragma unroll
            for (int ni = 0; ni < 2; ++ni)
                acc[mi][ni] = __builtin_amdgcn_wmma_f32_16x16x32_f16(
                    false, af[mi], false, bf[ni], (short)0, acc[mi][ni], false, false);
        __syncthreads();
    }

    // ---- epilogue: C/D layout lane = 16*hi + n, vgpr r -> m = r + 8*hi ----
#pragma unroll
    for (int mi = 0; mi < 2; ++mi) {
#pragma unroll
        for (int ni = 0; ni < 2; ++ni) {
#pragma unroll
            for (int r = 0; r < 8; ++r) {
                int gm = m0 + wm * 32 + mi * 16 + r + (hi << 3);
                int gn = n0 + wn * 32 + ni * 16 + lm;
                if (gm < M && gn < N) {
                    float v = acc[mi][ni][r];
                    if (biasMode == 1)      v += bias[gn];
                    else if (biasMode == 2) v += bias[gm];
                    if (resid) v += resid[(long)gm * ldc + gn];
                    if (act == 1) v = fmaxf(v, 0.f);
                    C[(long)gm * ldc + gn] = v;
                }
            }
        }
    }
}

// --------------------------- elementwise kernels ---------------------------

__device__ __forceinline__ float pos_embed(int pos, int c)
{
    const float kf = -9.210340371976184f / 256.0f;  // -ln(10000)/C
    float dv  = __expf((float)(c & ~1) * kf);
    float ang = (float)pos * dv;
    return (c & 1) ? __cosf(ang) : __sinf(ang);
}

// grouped conv1d, kernel 3, pad 1, optional relu. in[b,Cin,T] -> out[b,Cout,T]
__global__ void conv1d_grouped(const float* __restrict__ in, const float* __restrict__ w,
                               const float* __restrict__ bias, float* __restrict__ out,
                               int Bn, int Cin, int Cout, int T, int groups, int relu)
{
    int idx = blockIdx.x * blockDim.x + threadIdx.x;
    int total = Bn * Cout * T;
    if (idx >= total) return;
    int t  = idx % T;
    int co = (idx / T) % Cout;
    int b  = idx / (T * Cout);
    int cpg = Cin / groups, opg = Cout / groups;
    int g = co / opg;
    const float* wp = w + (long)co * cpg * 3;
    const float* ip = in + ((long)b * Cin + g * cpg) * T;
    float acc = bias[co];
    for (int ci = 0; ci < cpg; ++ci) {
        const float* ir = ip + (long)ci * T;
        const float* wr = wp + ci * 3;
        if (t > 0)     acc += ir[t - 1] * wr[0];
        acc += ir[t] * wr[1];
        if (t + 1 < T) acc += ir[t + 1] * wr[2];
    }
    if (relu) acc = fmaxf(acc, 0.f);
    out[idx] = acc;
}

// base[b,c,t] -> xg[(t*2+b)*256+c] (+PE), optional time flip of source
__global__ void build_global_seq(const float* __restrict__ base, float* __restrict__ xg, int flip)
{
    int idx = blockIdx.x * blockDim.x + threadIdx.x;
    if (idx >= T_SCALE * 2 * CCH) return;
    int c = idx % CCH;
    int b = (idx / CCH) % 2;
    int t = idx / (2 * CCH);
    int ts = flip ? (T_SCALE - 1 - t) : t;
    xg[idx] = base[((long)b * CCH + c) * T_SCALE + ts] + pos_embed(t, c);
}

// base[b,c,l*3+gr] -> xl[(l*6 + gr*2 + b)*256 + c] (+PE over l)
__global__ void build_local_seq(const float* __restrict__ base, float* __restrict__ xl)
{
    int idx = blockIdx.x * blockDim.x + threadIdx.x;
    if (idx >= 33 * 6 * CCH) return;
    int c = idx % CCH;
    int w = (idx / CCH) % 6;
    int l = idx / (6 * CCH);
    int gr = w >> 1, b = w & 1;
    int t = l * 3 + gr;
    xl[idx] = base[((long)b * CCH + c) * T_SCALE + t] + pos_embed(l, c);
}

// base2[b,c,t] = local(featL) + f1[t] + f2[T-1-t]
__global__ void compose_base2(const float* __restrict__ featL, const float* __restrict__ f1,
                              const float* __restrict__ f2, float* __restrict__ base2)
{
    int idx = blockIdx.x * blockDim.x + threadIdx.x;
    if (idx >= 2 * CCH * T_SCALE) return;
    int t = idx % T_SCALE;
    int c = (idx / T_SCALE) % CCH;
    int b = idx / (T_SCALE * CCH);
    int l = t / 3, gr = t % 3;
    float lv = featL[((long)(l * 6 + gr * 2 + b)) * CCH + c];
    float g1 = f1[((long)(t * 2 + b)) * CCH + c];
    float g2 = f2[((long)((T_SCALE - 1 - t) * 2 + b)) * CCH + c];
    base2[idx] = lv + g1 + g2;
}

// one block per (b, h, q); scores over k, softmax, weighted sum of V
__global__ __launch_bounds__(128)
void attention_kernel(const float* __restrict__ qkv, float* __restrict__ out, int T, int Bb)
{
    int q = blockIdx.x % T;
    int h = (blockIdx.x / T) % 8;
    int b = blockIdx.x / (T * 8);
    int tid = threadIdx.x;
    __shared__ float sc[128];
    __shared__ float red[128];
    const float scale = 0.17677669529663687f;  // 1/sqrt(32)
    float s = -1e30f;
    if (tid < T) {
        const float* qp = qkv + ((long)(q * Bb + b)) * 768 + h * 32;
        const float* kp = qkv + ((long)(tid * Bb + b)) * 768 + 256 + h * 32;
        float d = 0.f;
        for (int i = 0; i < 32; ++i) d += qp[i] * kp[i];
        s = d * scale;
    }
    sc[tid] = s;
    red[tid] = s;
    __syncthreads();
    for (int st = 64; st > 0; st >>= 1) {
        if (tid < st) red[tid] = fmaxf(red[tid], red[tid + st]);
        __syncthreads();
    }
    float mx = red[0];
    __syncthreads();
    float e = (tid < T) ? __expf(sc[tid] - mx) : 0.f;
    sc[tid] = e;
    red[tid] = e;
    __syncthreads();
    for (int st = 64; st > 0; st >>= 1) {
        if (tid < st) red[tid] += red[tid + st];
        __syncthreads();
    }
    float inv = 1.f / red[0];
    __syncthreads();
    if (tid < 32) {
        float o = 0.f;
        for (int k = 0; k < T; ++k)
            o += sc[k] * qkv[((long)(k * Bb + b)) * 768 + 512 + h * 32 + tid];
        out[((long)(q * Bb + b)) * CCH + h * 32 + tid] = o * inv;
    }
}

// one block (256 threads) per row of [rows, 256]
__global__ __launch_bounds__(256)
void layernorm_kernel(const float* __restrict__ x, const float* __restrict__ g,
                      const float* __restrict__ b, float* __restrict__ out)
{
    int row = blockIdx.x;
    int c = threadIdx.x;
    float v = x[(long)row * CCH + c];
    __shared__ float sh[256];
    sh[c] = v;
    __syncthreads();
    for (int st = 128; st > 0; st >>= 1) {
        if (c < st) sh[c] += sh[c + st];
        __syncthreads();
    }
    float mu = sh[0] * (1.f / 256.f);
    __syncthreads();
    float d = v - mu;
    sh[c] = d * d;
    __syncthreads();
    for (int st = 128; st > 0; st >>= 1) {
        if (c < st) sh[c] += sh[c + st];
        __syncthreads();
    }
    float var = sh[0] * (1.f / 256.f);
    out[(long)row * CCH + c] = d * rsqrtf(var + 1e-5f) * g[c] + b[c];
}

// M2[(s*99+t)*9801 + n] = mask[t*313632 + s*9801 + n]
__global__ void repack_mask(const float* __restrict__ mask, float* __restrict__ M2)
{
    long idx = (long)blockIdx.x * blockDim.x + threadIdx.x;
    const long total = (long)NSB * T_SCALE * TT;
    if (idx >= total) return;
    int n = (int)(idx % TT);
    int k = (int)(idx / TT);
    int s = k / T_SCALE, t = k % T_SCALE;
    M2[idx] = mask[(long)t * MASK_COLS + (long)s * TT + n];
}

// im2col for 3x3 pad-1 conv on a single [256,99,99] image
__global__ void im2col3x3(const float* __restrict__ in, float* __restrict__ col)
{
    long idx = (long)blockIdx.x * blockDim.x + threadIdx.x;
    const long total = 2304L * TT;
    if (idx >= total) return;
    int n = (int)(idx % TT);
    int k = (int)(idx / TT);
    int ci = k / 9, r = k % 9;
    int dy = r / 3 - 1, dx = r % 3 - 1;
    int y = n / T_SCALE + dy, xx = n % T_SCALE + dx;
    float v = 0.f;
    if (y >= 0 && y < T_SCALE && xx >= 0 && xx < T_SCALE)
        v = in[(long)ci * TT + y * T_SCALE + xx];
    col[idx] = v;
}

// start1 head: sigmoid(conv1x1) over sbuf[b,c,t]
__global__ void start_head(const float* __restrict__ s, const float* __restrict__ w,
                           const float* __restrict__ b, float* __restrict__ out)
{
    int idx = blockIdx.x * blockDim.x + threadIdx.x;
    if (idx >= 2 * T_SCALE) return;
    int t = idx % T_SCALE, bb = idx / T_SCALE;
    float acc = b[0];
    for (int c = 0; c < CCH; ++c)
        acc += s[((long)bb * CCH + c) * T_SCALE + t] * w[c];
    out[idx] = 1.f / (1.f + __expf(-acc));
}

// cm head: 1x1 conv 256->2 + sigmoid
__global__ void p3_head(const float* __restrict__ P, const float* __restrict__ w,
                        const float* __restrict__ bias, float* __restrict__ out)
{
    int idx = blockIdx.x * blockDim.x + threadIdx.x;
    if (idx >= 2 * 2 * TT) return;
    int n = idx % TT;
    int o = (idx / TT) % 2;
    int b = idx / (2 * TT);
    const float* Pb = P + (long)b * CCH * TT;
    const float* wo = w + o * CCH;
    float acc = bias[o];
    for (int ci = 0; ci < CCH; ++ci) acc += Pb[(long)ci * TT + n] * wo[ci];
    out[(long)b * 2 * TT + (long)o * TT + n] = 1.f / (1.f + __expf(-acc));
}

// ------------------------------- host side ---------------------------------

static void launch_gemm(hipStream_t st,
                        const float* A, long sAm, long sAk,
                        const float* B, long sBk, long sBn,
                        float* C, long ldc, int M, int N, int K,
                        const float* bias, int biasMode,
                        const float* resid, int act,
                        int Z = 1, int zdiv = 1,
                        long aZr = 0, long aZq = 0, long bZr = 0, long bZq = 0,
                        long cZr = 0, long cZq = 0)
{
    dim3 grid((N + BN - 1) / BN, (M + BM - 1) / BM, Z);
    const bool ak1 = (sAk == 1), bk1 = (sBk == 1);
    if (ak1 && bk1)
        gemm_wmma_f16<1, 1><<<grid, 256, 0, st>>>(A, sAm, sAk, B, sBk, sBn, C, ldc, M, N, K,
            bias, biasMode, resid, act, zdiv, aZr, aZq, bZr, bZq, cZr, cZq);
    else if (ak1)
        gemm_wmma_f16<1, 0><<<grid, 256, 0, st>>>(A, sAm, sAk, B, sBk, sBn, C, ldc, M, N, K,
            bias, biasMode, resid, act, zdiv, aZr, aZq, bZr, bZq, cZr, cZq);
    else if (bk1)
        gemm_wmma_f16<0, 1><<<grid, 256, 0, st>>>(A, sAm, sAk, B, sBk, sBn, C, ldc, M, N, K,
            bias, biasMode, resid, act, zdiv, aZr, aZq, bZr, bZq, cZr, cZq);
    else
        gemm_wmma_f16<0, 0><<<grid, 256, 0, st>>>(A, sAm, sAk, B, sBk, sBn, C, ldc, M, N, K,
            bias, biasMode, resid, act, zdiv, aZr, aZq, bZr, bZq, cZr, cZq);
}

struct EncW {
    const float *qkv_w, *qkv_b, *o_w, *o_b, *l1_w, *l1_b, *l2_w, *l2_b;
    const float *n1_g, *n1_b, *n2_g, *n2_b, *nf_g, *nf_b;
};

static void run_encoder(hipStream_t st, float* X, const EncW& w, int nlayers,
                        int T, int Bb, float* QKV, float* ATT, float* TMP, float* H)
{
    const int rows = T * Bb;  // 198 for both configurations
    for (int l = 0; l < nlayers; ++l) {
        const float* qw  = w.qkv_w + (long)l * 768 * 256;
        const float* qb  = w.qkv_b + (long)l * 768;
        const float* ow  = w.o_w   + (long)l * 256 * 256;
        const float* ob  = w.o_b   + (long)l * 256;
        const float* w1  = w.l1_w  + (long)l * 1024 * 256;
        const float* b1  = w.l1_b  + (long)l * 1024;
        const float* w2  = w.l2_w  + (long)l * 256 * 1024;
        const float* b2  = w.l2_b  + (long)l * 256;

        // QKV = X @ qkv_w^T + b
        launch_gemm(st, X, 256, 1, qw, 1, 256, QKV, 768, rows, 768, 256, qb, 1, nullptr, 0);
        attention_kernel<<<Bb * 8 * T, 128, 0, st>>>(QKV, ATT, T, Bb);
        // TMP = ATT @ o_w^T + b + X (residual)
        launch_gemm(st, ATT, 256, 1, ow, 1, 256, TMP, 256, rows, 256, 256, ob, 1, X, 0);
        layernorm_kernel<<<rows, 256, 0, st>>>(TMP, w.n1_g + l * 256, w.n1_b + l * 256, X);
        // H = relu(X @ l1^T + b)
        launch_gemm(st, X, 256, 1, w1, 1, 256, H, 1024, rows, 1024, 256, b1, 1, nullptr, 1);
        // TMP = H @ l2^T + b + X
        launch_gemm(st, H, 1024, 1, w2, 1, 1024, TMP, 256, rows, 256, 1024, b2, 1, X, 0);
        layernorm_kernel<<<rows, 256, 0, st>>>(TMP, w.n2_g + l * 256, w.n2_b + l * 256, X);
    }
    layernorm_kernel<<<rows, 256, 0, st>>>(X, w.nf_g, w.nf_b, X);
}

static inline int cdiv(long n, int b) { return (int)((n + b - 1) / b); }

extern "C" void kernel_launch(void* const* d_in, const int* in_sizes, int n_in,
                              void* d_out, int out_size, void* d_ws, size_t ws_size,
                              hipStream_t stream)
{
    const float* x    = (const float*)d_in[0];
    const float* cb_w = (const float*)d_in[1];
    const float* cb_b = (const float*)d_in[2];
    EncW tg, tl;
    tg.qkv_w = (const float*)d_in[3];  tg.qkv_b = (const float*)d_in[4];
    tg.o_w   = (const float*)d_in[5];  tg.o_b   = (const float*)d_in[6];
    tg.l1_w  = (const float*)d_in[7];  tg.l1_b  = (const float*)d_in[8];
    tg.l2_w  = (const float*)d_in[9];  tg.l2_b  = (const float*)d_in[10];
    tg.n1_g  = (const float*)d_in[11]; tg.n1_b  = (const float*)d_in[12];
    tg.n2_g  = (const float*)d_in[13]; tg.n2_b  = (const float*)d_in[14];
    tg.nf_g  = (const float*)d_in[15]; tg.nf_b  = (const float*)d_in[16];
    tl.qkv_w = (const float*)d_in[17]; tl.qkv_b = (const float*)d_in[18];
    tl.o_w   = (const float*)d_in[19]; tl.o_b   = (const float*)d_in[20];
    tl.l1_w  = (const float*)d_in[21]; tl.l1_b  = (const float*)d_in[22];
    tl.l2_w  = (const float*)d_in[23]; tl.l2_b  = (const float*)d_in[24];
    tl.n1_g  = (const float*)d_in[25]; tl.n1_b  = (const float*)d_in[26];
    tl.n2_g  = (const float*)d_in[27]; tl.n2_b  = (const float*)d_in[28];
    tl.nf_g  = (const float*)d_in[29]; tl.nf_b  = (const float*)d_in[30];
    const float* s1_w = (const float*)d_in[31];
    const float* s1_b = (const float*)d_in[32];
    const float* s2_w = (const float*)d_in[33];
    const float* s2_b = (const float*)d_in[34];
    const float* c3_w = (const float*)d_in[35];
    const float* c3_b = (const float*)d_in[36];
    const float* p1_w = (const float*)d_in[37];
    const float* p1_b = (const float*)d_in[38];
    const float* p2_w = (const float*)d_in[39];
    const float* p2_b = (const float*)d_in[40];
    const float* p3_w = (const float*)d_in[41];
    const float* p3_b = (const float*)d_in[42];
    const float* mask = (const float*)d_in[43];
    float* out = (float*)d_out;
    (void)in_sizes; (void)n_in; (void)out_size; (void)ws_size;

    // workspace bump allocator (256B aligned)
    char* wsp = (char*)d_ws;
    auto alloc = [&](long elems) {
        float* p = (float*)wsp;
        wsp += ((elems * 4 + 255) / 256) * 256;
        return p;
    };
    const long SEQ  = 198L * 256;       // 50688
    float* base  = alloc(SEQ);
    float* XL    = alloc(SEQ);          // local encoder state / output
    float* XG1   = alloc(SEQ);          // global fwd pass
    float* XG2   = alloc(SEQ);          // global flipped pass
    float* base2 = alloc(SEQ);
    float* sbuf  = alloc(SEQ);
    float* QKV   = alloc(198L * 768);
    float* ATT   = alloc(SEQ);
    float* TMP   = alloc(SEQ);
    float* H     = alloc(198L * 1024);
    float* G     = alloc(2L * 256 * (NSB * T_SCALE));   // [b, co, (s,t)]
    float* M2    = alloc((long)NSB * T_SCALE * TT);     // 124 MB repacked mask
    float* P0    = alloc(2L * 256 * TT);                // 20 MB
    float* P1    = alloc(2L * 256 * TT);
    float* col   = alloc(2304L * TT);                   // 90 MB, one batch at a time

    // 1. base = relu(grouped conv1d(x))
    conv1d_grouped<<<cdiv(2L * 256 * T_SCALE, 256), 256, 0, stream>>>(
        x, cb_w, cb_b, base, 2, 400, 256, T_SCALE, 4, 1);

    // 2. local transformer (3 layers) on 33-long windows, batch 6
    build_local_seq<<<cdiv(SEQ, 256), 256, 0, stream>>>(base, XL);
    run_encoder(stream, XL, tl, 3, 33, 6, QKV, ATT, TMP, H);

    // 3. global bidirectional transformer (6 layers, shared weights)
    build_global_seq<<<cdiv(SEQ, 256), 256, 0, stream>>>(base, XG1, 0);
    run_encoder(stream, XG1, tg, 6, T_SCALE, 2, QKV, ATT, TMP, H);
    build_global_seq<<<cdiv(SEQ, 256), 256, 0, stream>>>(base, XG2, 1);
    run_encoder(stream, XG2, tg, 6, T_SCALE, 2, QKV, ATT, TMP, H);

    // 4. base2 = local + f1 + flip(f2)   -> [b,c,t]
    compose_base2<<<cdiv(SEQ, 256), 256, 0, stream>>>(XL, XG1, XG2, base2);

    // 5. start branch: s1 grouped conv + relu, then 1x1 sigmoid head
    conv1d_grouped<<<cdiv(SEQ, 256), 256, 0, stream>>>(
        base2, s1_w, s1_b, sbuf, 2, 256, 256, T_SCALE, 4, 1);
    start_head<<<cdiv(2 * T_SCALE, 256), 256, 0, stream>>>(sbuf, s2_w, s2_b,
                                                           out + 2L * 2 * TT);

    // 6. fused BM head stage 1: G[b,co,(s,t)] = sum_ci c3_w[co,ci,s]*base2[b,ci,t]
    //    batched over z = b*32 + s (zdiv = 32)
    launch_gemm(stream,
                c3_w, 256L * NSB, NSB,          // A(co,ci) for fixed s (base + s)
                base2, T_SCALE, 1,              // B(ci,t) per batch b
                G, (long)NSB * T_SCALE,         // C ld = 3168
                256, T_SCALE, 256,
                nullptr, 0, nullptr, 0,
                /*Z=*/64, /*zdiv=*/32,
                /*aZr=*/1, /*aZq=*/0,
                /*bZr=*/0, /*bZq=*/(long)256 * T_SCALE,
                /*cZr=*/T_SCALE, /*cZq=*/(long)256 * NSB * T_SCALE);

    // 7. repack mask into (s,t)-major K layout; stays resident in 192MB L2
    repack_mask<<<cdiv((long)NSB * T_SCALE * TT, 256), 256, 0, stream>>>(mask, M2);

    // 8. stage 2: P0[b] = relu( G[b] (256x3168) x M2 (3168x9801) + c3_b )
    launch_gemm(stream,
                G, (long)NSB * T_SCALE, 1,
                M2, TT, 1,
                P0, TT,
                256, TT, NSB * T_SCALE,
                c3_b, 2, nullptr, 1,
                /*Z=*/2, /*zdiv=*/1,
                0, (long)256 * NSB * T_SCALE,
                0, 0,
                0, (long)256 * TT);

    // 9. p1: 3x3 conv via im2col + WMMA GEMM (per batch, reusing col buffer)
    for (int b = 0; b < 2; ++b) {
        im2col3x3<<<cdiv(2304L * TT, 256), 256, 0, stream>>>(P0 + (long)b * 256 * TT, col);
        launch_gemm(stream, p1_w, 2304, 1, col, TT, 1,
                    P1 + (long)b * 256 * TT, TT, 256, TT, 2304, p1_b, 2, nullptr, 1);
    }
    // 10. p2
    for (int b = 0; b < 2; ++b) {
        im2col3x3<<<cdiv(2304L * TT, 256), 256, 0, stream>>>(P1 + (long)b * 256 * TT, col);
        launch_gemm(stream, p2_w, 2304, 1, col, TT, 1,
                    P0 + (long)b * 256 * TT, TT, 256, TT, 2304, p2_b, 2, nullptr, 1);
    }

    // 11. cm = sigmoid(p3 1x1 conv)  -> out[0 .. 39203]
    p3_head<<<cdiv(2L * 2 * TT, 256), 256, 0, stream>>>(P0, p3_w, p3_b, out);
}
